// Net_87359634800659
// MI455X (gfx1250) — matrix operand order repaired and checked by reading, not compiled, not run on hardware
//
#include <hip/hip_runtime.h>
#include <math.h>

// ---------------------------------------------------------------------------
// 2-layer GAT for MI455X (gfx1250, wave32).
// GEMMs: v_wmma_f32_16x16x32_bf16 with hi/lo split-bf16 (3 WMMA per K-step)
// for ~fp32 accuracy. LDS staging uses CDNA5 GLOBAL_LOAD_ASYNC_TO_LDS_B128
// (ASYNCcnt-tracked, double-buffered, last step peeled) so WMMA overlaps
// global latency. Graph phases use f32 atomics + wave shuffles.
// ---------------------------------------------------------------------------

#define NEG_SLOPE 0.2f

typedef unsigned short u16;
typedef __attribute__((ext_vector_type(16))) __bf16        v16bf;
typedef __attribute__((ext_vector_type(8)))  float         v8f;
typedef __attribute__((ext_vector_type(4)))  unsigned int  u32x4;

struct Frag32B { u32x4 a, b; };

__device__ __forceinline__ u16 f32_to_bf16_rne(float f) {
    unsigned int u = __float_as_uint(f);
    unsigned int r = u + 0x7FFFu + ((u >> 16) & 1u);
    return (u16)(r >> 16);
}
__device__ __forceinline__ float bf16_to_f32(u16 h) {
    return __uint_as_float(((unsigned int)h) << 16);
}

// CDNA5 async global->LDS 16B transfer (per-lane), tracked by ASYNCcnt.
__device__ __forceinline__ void async_ld16(void* lds, const void* gp) {
    unsigned loff = (unsigned)(size_t)lds;   // flat addr[31:0] == LDS offset
    asm volatile("global_load_async_to_lds_b128 %0, %1, off"
                 :: "v"(loff), "v"(gp)
                 : "memory");
}
__device__ __forceinline__ void wait_async0() {
    asm volatile("s_wait_asynccnt 0x0" ::: "memory");
}
__device__ __forceinline__ void wait_async3() {
    asm volatile("s_wait_asynccnt 0x3" ::: "memory");
}

// ---------------- utility kernels ----------------

__global__ void fill_f32(float* __restrict__ p, float v, long long n) {
    long long i = (long long)blockIdx.x * blockDim.x + threadIdx.x;
    if (i < n) p[i] = v;
}

// f32 -> bf16 hi/lo split (row-major, elementwise)
__global__ void split_bf16(const float* __restrict__ x, u16* __restrict__ hi,
                           u16* __restrict__ lo, long long n) {
    long long i = (long long)blockIdx.x * blockDim.x + threadIdx.x;
    if (i >= n) return;
    float v = x[i];
    u16 h = f32_to_bf16_rne(v);
    hi[i] = h;
    lo[i] = f32_to_bf16_rne(v - bf16_to_f32(h));
}

// W (K x N, row-major) -> transposed bf16 hi/lo (N x K, row-major)
__global__ void split_bf16_t(const float* __restrict__ W, u16* __restrict__ hiT,
                             u16* __restrict__ loT, int K, int N) {
    long long i = (long long)blockIdx.x * blockDim.x + threadIdx.x;
    if (i >= (long long)K * N) return;
    int k = (int)(i / N), n = (int)(i % N);
    float v = W[i];
    u16 h = f32_to_bf16_rne(v);
    hiT[(long long)n * K + k] = h;
    loT[(long long)n * K + k] = f32_to_bf16_rne(v - bf16_to_f32(h));
}

// ---------------- WMMA GEMM ----------------
// C[M,N] = A[M,K] @ B[K,N]; A given as bf16 hi/lo (M x K), B given transposed
// bf16 hi/lo (N x K). N % 64 == 0, K % 32 == 0 (K >= 64), M arbitrary.

#define BM 32
#define BN 64
#define KC 32
#define LDA (KC + 8)   // ushort stride, 80B rows keep 16B alignment

__global__ __launch_bounds__(256) void gemm_bf16x3(
    const u16* __restrict__ Ah, const u16* __restrict__ Al,
    const u16* __restrict__ Bh, const u16* __restrict__ Bl,
    float* __restrict__ C, int M, int N, int K)
{
    // double-buffered staging
    __shared__ __attribute__((aligned(16))) u16 sAh[2][BM][LDA];
    __shared__ __attribute__((aligned(16))) u16 sAl[2][BM][LDA];
    __shared__ __attribute__((aligned(16))) u16 sBh[2][BN][LDA];
    __shared__ __attribute__((aligned(16))) u16 sBl[2][BN][LDA];

    const int t        = threadIdx.x;
    const int blockRow = blockIdx.y;
    const int blockCol = blockIdx.x;

    // A staging: 256 x 16B chunks (128 hi + 128 lo) = 32 rows x 32 K x 2 mats.
    // Row index clamped (not predicated) so async transfers keep EXEC full;
    // duplicate rows only feed accumulator rows that are never stored.
    const int aMat  = t >> 7;
    const int aIdx  = t & 127;
    const int aRow  = aIdx >> 2;
    const int aOff  = (aIdx & 3) * 8;          // ushort offset
    int aGRow = blockRow * BM + aRow;
    if (aGRow > M - 1) aGRow = M - 1;
    const u16* Asrc = aMat ? Al : Ah;

    // B staging: each thread moves one 16B chunk of hi and one of lo
    const int       bRow  = t >> 2;             // 0..63
    const int       bOff  = (t & 3) * 8;
    const long long bGRow = (long long)blockCol * BN + bRow;

    // per-thread streaming pointers, advanced by KC each K-step
    const u16* aPtr  = Asrc + (long long)aGRow * K + aOff;
    const u16* bhPtr = Bh + bGRow * K + bOff;
    const u16* blPtr = Bl + bGRow * K + bOff;

    // wave tile assignment: 8 waves as 2 (M) x 4 (N) grid of 16x16 tiles
    const int lane = t & 31;
    const int lr   = lane & 15;
    const int hsel = lane >> 4;                 // 0 | 1
    const int wid  = t >> 5;
    const int tR   = (wid >> 2) * 16;
    const int tC   = (wid & 3) * 16;

    v8f acc = {};

    auto compute_tile = [&](int cur) {
        // A fragment per ISA 16-bit 16x32 layout:
        // lanes 0-15 row M=lr, K {0..7,16..23}; lanes 16-31 K {8..15,24..31}
        const u16* paH = &sAh[cur][tR + lr][0];
        const u16* paL = &sAl[cur][tR + lr][0];
        // B fragment per ISA 16-bit Kx16 layout:
        // lanes 0-15 col N=lr, K {0..7,16..23}; lanes 16-31 K {8..15,24..31}
        const u16* pbH = &sBh[cur][tC + lr][0];
        const u16* pbL = &sBl[cur][tC + lr][0];

        Frag32B f;
        f.a = *(const u32x4*)(paH + hsel * 8);
        f.b = *(const u32x4*)(paH + 16 + hsel * 8);
        v16bf A_hi = __builtin_bit_cast(v16bf, f);
        f.a = *(const u32x4*)(paL + hsel * 8);
        f.b = *(const u32x4*)(paL + 16 + hsel * 8);
        v16bf A_lo = __builtin_bit_cast(v16bf, f);
        f.a = *(const u32x4*)(pbH + hsel * 8);
        f.b = *(const u32x4*)(pbH + 16 + hsel * 8);
        v16bf B_hi = __builtin_bit_cast(v16bf, f);
        f.a = *(const u32x4*)(pbL + hsel * 8);
        f.b = *(const u32x4*)(pbL + 16 + hsel * 8);
        v16bf B_lo = __builtin_bit_cast(v16bf, f);

        acc = __builtin_amdgcn_wmma_f32_16x16x32_bf16(
                  false, A_lo, false, B_hi, (short)0, acc, false, false);
        acc = __builtin_amdgcn_wmma_f32_16x16x32_bf16(
                  false, A_hi, false, B_lo, (short)0, acc, false, false);
        acc = __builtin_amdgcn_wmma_f32_16x16x32_bf16(
                  false, A_hi, false, B_hi, (short)0, acc, false, false);
    };

    const int T = K / KC;

    // prologue: async-stage tile 0 into buffer 0
    {
        u16* aDst = aMat ? &sAl[0][aRow][aOff] : &sAh[0][aRow][aOff];
        async_ld16(aDst, aPtr);
        async_ld16(&sBh[0][bRow][bOff], bhPtr);
        async_ld16(&sBl[0][bRow][bOff], blPtr);
    }

    // steady state: unconditionally issue next tile, wait for current group
    for (int i = 0; i < T - 1; ++i) {
        const int cur = i & 1;
        const int nxt = cur ^ 1;
        aPtr += KC; bhPtr += KC; blPtr += KC;
        u16* aDst = aMat ? &sAl[nxt][aRow][aOff] : &sAh[nxt][aRow][aOff];
        async_ld16(aDst, aPtr);
        async_ld16(&sBh[nxt][bRow][bOff], bhPtr);
        async_ld16(&sBl[nxt][bRow][bOff], blPtr);
        wait_async3();     // current tile's 3 transfers complete (FIFO)
        __syncthreads();   // visible to all waves
        compute_tile(cur);
        __syncthreads();   // all reads done before buffer reuse
    }

    // epilogue: last tile
    wait_async0();
    __syncthreads();
    compute_tile((T - 1) & 1);

    // C/D layout: lanes 0-15 reg r -> M = r; lanes 16-31 reg r -> M = 8+r
    const int n  = blockCol * BN + tC + lr;
    const int mb = blockRow * BM + tR + hsel * 8;
#pragma unroll
    for (int r = 0; r < 8; ++r) {
        int m = mb + r;
        if (m < M) C[(long long)m * N + n] = acc[r];
    }
}

// ---------------- attention scores: out[n,h] = dot(xs[n,h,:], att[h,:]) ----

__global__ void attn_score(const float* __restrict__ xs, const float* __restrict__ att,
                           float* __restrict__ out, int Nn, int H) {
    int node = blockIdx.x * (blockDim.x >> 5) + (threadIdx.x >> 5);
    int lane = threadIdx.x & 31;
    if (node >= Nn) return;
    const float* row = xs + (long long)node * H * 64;
    for (int h = 0; h < H; ++h) {
        float p = row[h * 64 + lane]      * att[h * 64 + lane]
                + row[h * 64 + lane + 32] * att[h * 64 + lane + 32];
#pragma unroll
        for (int off = 16; off > 0; off >>= 1) p += __shfl_xor(p, off, 32);
        if (lane == 0) out[(long long)node * H + h] = p;
    }
}

// ---------------- edge kernels ----------------

__device__ __forceinline__ void atomicMaxF32(float* addr, float val) {
    int iv = __float_as_int(val);
    if (iv >= 0) atomicMax((int*)addr, iv);
    else         atomicMin((unsigned int*)addr, (unsigned int)iv);
}

__global__ void edge_scores(const float* __restrict__ as_, const float* __restrict__ ad_,
                            const int* __restrict__ src, const int* __restrict__ dst,
                            float* __restrict__ e, float* m, long long EH, int H) {
    long long idx = (long long)blockIdx.x * blockDim.x + threadIdx.x;
    if (idx >= EH) return;
    int eid = (int)(idx / H), h = (int)(idx % H);
    int s = src[eid], d = dst[eid];
    float v = as_[(long long)s * H + h] + ad_[(long long)d * H + h];
    v = v > 0.f ? v : NEG_SLOPE * v;
    e[idx] = v;
    atomicMaxF32(&m[(long long)d * H + h], v);
}

__global__ void edge_exp(const float* e, const float* __restrict__ m,
                         const int* __restrict__ dst, float* ex, float* denom,
                         long long EH, int H) {
    long long idx = (long long)blockIdx.x * blockDim.x + threadIdx.x;
    if (idx >= EH) return;
    int eid = (int)(idx / H), h = (int)(idx % H);
    int d = dst[eid];
    float mm = m[(long long)d * H + h];
    if (!isfinite(mm)) mm = 0.f;
    float v = expf(e[idx] - mm);
    ex[idx] = v;
    atomicAdd(&denom[(long long)d * H + h], v);
}

// out[dst, :] += xs[src, :] * alpha  (one wave per edge)
__global__ void edge_aggregate(const float* __restrict__ xs, const float* __restrict__ ex,
                               const float* __restrict__ denom,
                               const int* __restrict__ src, const int* __restrict__ dst,
                               float* out, int E, int H) {
    int eid  = blockIdx.x * (blockDim.x >> 5) + (threadIdx.x >> 5);
    int lane = threadIdx.x & 31;
    if (eid >= E) return;
    int s = src[eid], d = dst[eid];
    float alpha[4];
    for (int h = 0; h < H; ++h)
        alpha[h] = ex[(long long)eid * H + h] / (denom[(long long)d * H + h] + 1e-16f);
    int F = H * 64;
    for (int i = lane; i < F; i += 32) {
        atomicAdd(&out[(long long)d * F + i],
                  xs[(long long)s * F + i] * alpha[i >> 6]);
    }
}

// ---------------- post: out = aggr + bias + skip + skip_b (+ elu) ----------

__global__ void post_kernel(const float* __restrict__ aggr, const float* __restrict__ skip,
                            const float* __restrict__ bias, const float* __restrict__ skipb,
                            float* __restrict__ out, long long total, int N, int do_elu) {
    long long i = (long long)blockIdx.x * blockDim.x + threadIdx.x;
    if (i >= total) return;
    int col = (int)(i % N);
    float v = aggr[i] + bias[col] + skip[i] + skipb[col];
    if (do_elu) v = v > 0.f ? v : (expf(v) - 1.f);
    out[i] = v;
}

// ---------------- log_softmax over rows of 64 (one wave per row) -----------

__global__ void log_softmax64(float* __restrict__ o, int M) {
    int row  = blockIdx.x * (blockDim.x >> 5) + (threadIdx.x >> 5);
    int lane = threadIdx.x & 31;
    if (row >= M) return;
    float x0 = o[(long long)row * 64 + lane];
    float x1 = o[(long long)row * 64 + lane + 32];
    float mx = fmaxf(x0, x1);
#pragma unroll
    for (int off = 16; off > 0; off >>= 1) mx = fmaxf(mx, __shfl_xor(mx, off, 32));
    float s = expf(x0 - mx) + expf(x1 - mx);
#pragma unroll
    for (int off = 16; off > 0; off >>= 1) s += __shfl_xor(s, off, 32);
    float lse = mx + logf(s);
    o[(long long)row * 64 + lane]      = x0 - lse;
    o[(long long)row * 64 + lane + 32] = x1 - lse;
}

// ---------------- host-side orchestration ----------------

#define CEILDIV(a, b) (((a) + (b) - 1) / (b))

extern "C" void kernel_launch(void* const* d_in, const int* in_sizes, int n_in,
                              void* d_out, int out_size, void* d_ws, size_t ws_size,
                              hipStream_t stream) {
    (void)n_in; (void)out_size; (void)ws_size;

    const float* x     = (const float*)d_in[0];
    const int*   src1  = (const int*)d_in[1];
    const int*   dst1  = (const int*)d_in[2];
    const int*   src2  = (const int*)d_in[3];
    const int*   dst2  = (const int*)d_in[4];
    const float* W1s   = (const float*)d_in[7];
    const float* W1d   = (const float*)d_in[8];
    const float* att1s = (const float*)d_in[9];
    const float* att1d = (const float*)d_in[10];
    const float* bias1 = (const float*)d_in[11];
    const float* sk1W  = (const float*)d_in[12];
    const float* sk1b  = (const float*)d_in[13];
    const float* W2s   = (const float*)d_in[14];
    const float* W2d   = (const float*)d_in[15];
    const float* att2s = (const float*)d_in[16];
    const float* att2d = (const float*)d_in[17];
    const float* bias2 = (const float*)d_in[18];
    const float* sk2W  = (const float*)d_in[19];
    const float* sk2b  = (const float*)d_in[20];

    const int IN = 256, F1 = 256, F2 = 64, H1 = 4, H2 = 1;
    const int N0  = in_sizes[0] / IN;      // 100000
    const int E1  = in_sizes[1];           // 400000
    const int E2  = in_sizes[3];           // 100000
    const int NT1 = 25000, NT2 = 5000;     // reference constants

    char* wp = (char*)d_ws;
    auto alloc = [&](size_t bytes) -> char* {
        char* p = wp; wp += (bytes + 255) & ~(size_t)255; return p;
    };

    float* xs1   = (float*)alloc((size_t)N0  * F1 * 4);
    float* xd1   = (float*)alloc((size_t)NT1 * F1 * 4);
    float* sk1   = (float*)alloc((size_t)NT1 * F1 * 4);
    float* hbuf  = (float*)alloc((size_t)NT1 * F1 * 4);
    float* aggr1 = (float*)alloc((size_t)NT1 * F1 * 4);
    float* xs2   = (float*)alloc((size_t)NT1 * F2 * 4);
    float* xd2   = (float*)alloc((size_t)NT2 * F2 * 4);
    float* sk2   = (float*)alloc((size_t)NT2 * F2 * 4);
    float* aggr2 = (float*)alloc((size_t)NT2 * F2 * 4);
    float* as1   = (float*)alloc((size_t)N0  * H1 * 4);
    float* ad1   = (float*)alloc((size_t)NT1 * H1 * 4);
    float* m1    = (float*)alloc((size_t)NT1 * H1 * 4);
    float* dn1   = (float*)alloc((size_t)NT1 * H1 * 4);
    float* ex1   = (float*)alloc((size_t)E1  * H1 * 4);
    float* as2   = (float*)alloc((size_t)NT1 * 4);
    float* ad2   = (float*)alloc((size_t)NT2 * 4);
    float* m2    = (float*)alloc((size_t)NT2 * 4);
    float* dn2   = (float*)alloc((size_t)NT2 * 4);
    float* ex2   = (float*)alloc((size_t)E2  * 4);

    u16* xh    = (u16*)alloc((size_t)N0  * IN * 2);
    u16* xl    = (u16*)alloc((size_t)N0  * IN * 2);
    u16* hh    = (u16*)alloc((size_t)NT1 * F1 * 2);
    u16* hl    = (u16*)alloc((size_t)NT1 * F1 * 2);
    u16* w1s_h = (u16*)alloc((size_t)IN * F1 * 2);
    u16* w1s_l = (u16*)alloc((size_t)IN * F1 * 2);
    u16* w1d_h = (u16*)alloc((size_t)IN * F1 * 2);
    u16* w1d_l = (u16*)alloc((size_t)IN * F1 * 2);
    u16* s1w_h = (u16*)alloc((size_t)IN * F1 * 2);
    u16* s1w_l = (u16*)alloc((size_t)IN * F1 * 2);
    u16* w2s_h = (u16*)alloc((size_t)F1 * F2 * 2);
    u16* w2s_l = (u16*)alloc((size_t)F1 * F2 * 2);
    u16* w2d_h = (u16*)alloc((size_t)F1 * F2 * 2);
    u16* w2d_l = (u16*)alloc((size_t)F1 * F2 * 2);
    u16* s2w_h = (u16*)alloc((size_t)F1 * F2 * 2);
    u16* s2w_l = (u16*)alloc((size_t)F1 * F2 * 2);

    const int TB = 256;

    // weights -> transposed bf16 hi/lo
    split_bf16_t<<<CEILDIV(IN * F1, TB), TB, 0, stream>>>(W1s, w1s_h, w1s_l, IN, F1);
    split_bf16_t<<<CEILDIV(IN * F1, TB), TB, 0, stream>>>(W1d, w1d_h, w1d_l, IN, F1);
    split_bf16_t<<<CEILDIV(IN * F1, TB), TB, 0, stream>>>(sk1W, s1w_h, s1w_l, IN, F1);
    split_bf16_t<<<CEILDIV(F1 * F2, TB), TB, 0, stream>>>(W2s, w2s_h, w2s_l, F1, F2);
    split_bf16_t<<<CEILDIV(F1 * F2, TB), TB, 0, stream>>>(W2d, w2d_h, w2d_l, F1, F2);
    split_bf16_t<<<CEILDIV(F1 * F2, TB), TB, 0, stream>>>(sk2W, s2w_h, s2w_l, F1, F2);

    // activations -> bf16 hi/lo
    long long nx = (long long)N0 * IN;
    split_bf16<<<(unsigned)CEILDIV(nx, TB), TB, 0, stream>>>(x, xh, xl, nx);

    // layer-1 GEMMs
    gemm_bf16x3<<<dim3(F1 / BN, CEILDIV(N0, BM)), TB, 0, stream>>>(
        xh, xl, w1s_h, w1s_l, xs1, N0, F1, IN);
    gemm_bf16x3<<<dim3(F1 / BN, CEILDIV(NT1, BM)), TB, 0, stream>>>(
        xh, xl, w1d_h, w1d_l, xd1, NT1, F1, IN);
    gemm_bf16x3<<<dim3(F1 / BN, CEILDIV(NT1, BM)), TB, 0, stream>>>(
        xh, xl, s1w_h, s1w_l, sk1, NT1, F1, IN);

    // layer-1 attention
    attn_score<<<CEILDIV(N0, 8), TB, 0, stream>>>(xs1, att1s, as1, N0, H1);
    attn_score<<<CEILDIV(NT1, 8), TB, 0, stream>>>(xd1, att1d, ad1, NT1, H1);

    fill_f32<<<CEILDIV(NT1 * H1, TB), TB, 0, stream>>>(m1, -INFINITY, (long long)NT1 * H1);
    fill_f32<<<CEILDIV(NT1 * H1, TB), TB, 0, stream>>>(dn1, 0.f, (long long)NT1 * H1);
    fill_f32<<<CEILDIV(NT1 * F1, TB), TB, 0, stream>>>(aggr1, 0.f, (long long)NT1 * F1);

    long long eh1 = (long long)E1 * H1;
    edge_scores<<<(unsigned)CEILDIV(eh1, TB), TB, 0, stream>>>(as1, ad1, src1, dst1, ex1, m1, eh1, H1);
    edge_exp<<<(unsigned)CEILDIV(eh1, TB), TB, 0, stream>>>(ex1, m1, dst1, ex1, dn1, eh1, H1);
    edge_aggregate<<<CEILDIV(E1, 8), TB, 0, stream>>>(xs1, ex1, dn1, src1, dst1, aggr1, E1, H1);

    post_kernel<<<CEILDIV(NT1 * F1, TB), TB, 0, stream>>>(
        aggr1, sk1, bias1, sk1b, hbuf, (long long)NT1 * F1, F1, 1);

    // layer-2
    long long nh = (long long)NT1 * F1;
    split_bf16<<<(unsigned)CEILDIV(nh, TB), TB, 0, stream>>>(hbuf, hh, hl, nh);

    gemm_bf16x3<<<dim3(F2 / BN, CEILDIV(NT1, BM)), TB, 0, stream>>>(
        hh, hl, w2s_h, w2s_l, xs2, NT1, F2, F1);
    gemm_bf16x3<<<dim3(F2 / BN, CEILDIV(NT2, BM)), TB, 0, stream>>>(
        hh, hl, w2d_h, w2d_l, xd2, NT2, F2, F1);
    gemm_bf16x3<<<dim3(F2 / BN, CEILDIV(NT2, BM)), TB, 0, stream>>>(
        hh, hl, s2w_h, s2w_l, sk2, NT2, F2, F1);

    attn_score<<<CEILDIV(NT1, 8), TB, 0, stream>>>(xs2, att2s, as2, NT1, H2);
    attn_score<<<CEILDIV(NT2, 8), TB, 0, stream>>>(xd2, att2d, ad2, NT2, H2);

    fill_f32<<<CEILDIV(NT2, TB), TB, 0, stream>>>(m2, -INFINITY, (long long)NT2);
    fill_f32<<<CEILDIV(NT2, TB), TB, 0, stream>>>(dn2, 0.f, (long long)NT2);
    fill_f32<<<CEILDIV(NT2 * F2, TB), TB, 0, stream>>>(aggr2, 0.f, (long long)NT2 * F2);

    edge_scores<<<CEILDIV(E2, TB), TB, 0, stream>>>(as2, ad2, src2, dst2, ex2, m2, (long long)E2, H2);
    edge_exp<<<CEILDIV(E2, TB), TB, 0, stream>>>(ex2, m2, dst2, ex2, dn2, (long long)E2, H2);
    edge_aggregate<<<CEILDIV(E2, 8), TB, 0, stream>>>(xs2, ex2, dn2, src2, dst2, aggr2, E2, H2);

    float* out = (float*)d_out;
    post_kernel<<<CEILDIV(NT2 * F2, TB), TB, 0, stream>>>(
        aggr2, sk2, bias2, sk2b, out, (long long)NT2 * F2, F2, 0);
    log_softmax64<<<CEILDIV(NT2, 8), TB, 0, stream>>>(out, NT2);
}
